// TCHead_29463475651412
// MI455X (gfx1250) — matrix operand order repaired and checked by reading, not compiled, not run on hardware
//
#include <hip/hip_runtime.h>

// MI455X / gfx1250, wave32. Conv head as implicit-GEMM via
// V_WMMA_F32_16X16X32_F16 (M=16 out-channels, N=16 pixels).
// GEMM K is reordered into (c,kh)-chunks of 16 (kw 0..14 + pad) so that each
// lane's per-k-step B fragment is 16 consecutive f16 in the LDS patch row.
// A fragments are pre-packed (K0) into a contiguous-k A layout (k = 16*hi + s) in global ws.
//
// ws layout (bytes):
//   [0)              xm  : 64*16*16384 f32          = 67,108,864
//   [67108864)       h1  : 64*16*16384 f32          = 67,108,864
//   [134217728)      Ahead: 15*32*16 f16            = 15,360
//   [134233088)      Asq1 : 120*32*16 f16           = 122,880
// total required ws_size >= 134,355,968 bytes.

typedef _Float16 f16;
typedef _Float16 v16h __attribute__((ext_vector_type(16)));
typedef float    v8f  __attribute__((ext_vector_type(8)));

#define T_DIM 16
#define B_DIM 4
#define NTC   5
#define CIN   2
#define CH    16
#define HW    128
#define HWHW  (HW*HW)

#define KS1   15     // head: 30 chunks -> 15 k-steps
#define KS2   120    // sq1: 240 chunks -> 120 k-steps

// ---------------------------------------------------------------------------
// K0: pack head_w and sq1_w into A-fragment layout (contiguous k = 16*lhi + s, chunked K ordering).
// fragment-lane g holds 16 f16: element 2v+h = weight at k = ks*32+koff+h,
// koff = lhi*16 + 2v;  k -> chunk=k>>4 (c,kh), kw=k&15.
// ---------------------------------------------------------------------------
__global__ void __launch_bounds__(256) k0_pack(
    const float* __restrict__ head_w, const float* __restrict__ sq1_w,
    f16* __restrict__ Ahead, f16* __restrict__ Asq1) {
  int g = blockIdx.x * 256 + threadIdx.x;
  if (g < KS1 * 32) {
    int ks = g >> 5, L = g & 31;
    int m = L & 15, lhi = (L >> 4) & 1;
    for (int v = 0; v < 8; ++v) {
      int koff = lhi * 16 + 2 * v;
      for (int h = 0; h < 2; ++h) {
        int k = ks * 32 + koff + h;
        int chunk = k >> 4, kw = k & 15;
        f16 w = (f16)0.f;
        if (kw < 15) {                       // chunk = c*15 + kh, c in 0..1
          int c = chunk >= 15 ? 1 : 0;
          int kh = chunk - c * 15;
          w = (f16)head_w[((m * CIN + c) * 15 + kh) * 15 + kw];
        }
        Ahead[(size_t)g * 16 + 2 * v + h] = w;
      }
    }
  } else if (g < KS1 * 32 + KS2 * 32) {
    int gg = g - KS1 * 32;
    int ks = gg >> 5, L = gg & 31;
    int m = L & 15, lhi = (L >> 4) & 1;
    for (int v = 0; v < 8; ++v) {
      int koff = lhi * 16 + 2 * v;
      for (int h = 0; h < 2; ++h) {
        int k = ks * 32 + koff + h;
        int chunk = k >> 4, kw = k & 15;     // chunk in 0..239
        f16 w = (f16)0.f;
        if (kw < 15) {
          int c = chunk / 15;                // 0..15
          int kh = chunk - c * 15;
          w = (f16)sq1_w[((m * CH + c) * 15 + kh) * 15 + kw];
        }
        Asq1[(size_t)gg * 16 + 2 * v + h] = w;
      }
    }
  }
}

// ---------------------------------------------------------------------------
// K1: head conv (per TC slice) + leaky-integrator scan over T + TC mean -> xm
// grid (128 tiles, 4 batches), block 256 (8 waves; wave w owns row py0+w).
// ---------------------------------------------------------------------------
__global__ void __launch_bounds__(256) k1_head_scan(
    const float* __restrict__ x, const f16* __restrict__ Ahead,
    const float* __restrict__ mtc, float* __restrict__ xm) {
  __shared__ __align__(32) f16 As[KS1 * 32 * 16];  // 15 KB packed fragments
  __shared__ f16 patch[CIN][22][32];               // input tile, 2.75 KB

  const int tid  = threadIdx.x;
  const int wid  = tid >> 5;
  const int lane = tid & 31;
  const int tile = blockIdx.x;       // 0..127
  const int b    = blockIdx.y;       // 0..3
  const int py0  = (tile >> 3) * 8;
  const int px0  = (tile & 7) * 16;
  const int hi   = (lane >> 4) & 1;
  const int n    = lane & 15;

  // straight vector copy of packed fragments: global -> LDS (no gather)
  {
    const uint4* s = reinterpret_cast<const uint4*>(Ahead);
    uint4*       d = reinterpret_cast<uint4*>(As);
    for (int i = tid; i < KS1 * 32 * 16 * 2 / 16; i += 256) d[i] = s[i];
  }

  // log-spaced time constants: tau = mtc * sqrt(2)^(tc-2)
  float alpha[NTC];
  const float m_tc = mtc[0];
  for (int tc = 0; tc < NTC; ++tc) {
    float tau = m_tc * exp2f(0.5f * (float)(tc - 2));
    alpha[tc] = __expf(-1.0f / tau);
  }

  v8f st[NTC];
  for (int tc = 0; tc < NTC; ++tc)
    for (int i = 0; i < 8; ++i) st[tc][i] = 0.f;

  const int py = py0 + wid;
  const int px = px0 + n;

  for (int t = 0; t < T_DIM; ++t) {
    v8f xacc; for (int i = 0; i < 8; ++i) xacc[i] = 0.f;
    for (int tc = 0; tc < NTC; ++tc) {
      __syncthreads();                       // previous patch fully consumed
      for (int e = tid; e < CIN * 22 * 32; e += 256) {
        int cc = e & 31;
        int rr = e >> 5;                     // 0..43
        int c  = rr >= 22 ? 1 : 0;
        int r  = rr - c * 22;
        int iy = py0 + r - 7, ix = px0 + cc - 7;
        float val = 0.f;
        if (iy >= 0 && iy < HW && ix >= 0 && ix < HW)
          val = x[((((size_t)t * B_DIM + b) * (NTC * CIN) + (tc * CIN + c)) * HW + iy) * HW + ix];
        patch[c][r][cc] = (f16)val;
      }
      __syncthreads();

      v8f acc; for (int i = 0; i < 8; ++i) acc[i] = 0.f;
      for (int ks = 0; ks < KS1; ++ks) {
        v16h a = reinterpret_cast<const v16h*>(As)[ks * 32 + lane];
        int chunk = ks * 2 + hi;             // 0..29
        int c  = chunk >= 15 ? 1 : 0;
        int kh = chunk - c * 15;
        const f16* base = &patch[c][wid + kh][n];
        v16h bf;
#pragma unroll
        for (int u = 0; u < 15; ++u) bf[u] = base[u];  // 15 consecutive f16
        bf[15] = (f16)0.f;                             // kw pad
        acc = __builtin_amdgcn_wmma_f32_16x16x32_f16(
            false, a, false, bf, (short)0, acc, false, false);
      }
      float al = alpha[tc];
      st[tc] = st[tc] * al + acc * (1.0f - al);   // leaky integrator
      xacc += st[tc];
    }
    v8f mo = xacc * 0.2f;                         // mean over 5 TCs
    for (int r = 0; r < 8; ++r) {
      int ch = r + hi * 8;
      xm[(((size_t)t * B_DIM + b) * CH + ch) * HWHW + py * HW + px] = mo[r];
    }
  }
}

// ---------------------------------------------------------------------------
// K2: sq1 conv (16->16, 15x15): xm -> h1.  grid (128 tiles, 64 images).
// A fragments read straight from L2-resident packed table (32B/lane, coalesced).
// ---------------------------------------------------------------------------
__global__ void __launch_bounds__(256) k2_sq1(
    const float* __restrict__ xm, const f16* __restrict__ Asq1,
    float* __restrict__ h1) {
  __shared__ f16 patch[CH][22][32];              // 22 KB input tile

  const int tid  = threadIdx.x;
  const int wid  = tid >> 5;
  const int lane = tid & 31;
  const int tile = blockIdx.x;
  const int img  = blockIdx.y;                   // t*B+b, 0..63
  const int py0  = (tile >> 3) * 8;
  const int px0  = (tile & 7) * 16;
  const int hi   = (lane >> 4) & 1;
  const int n    = lane & 15;

  for (int e = tid; e < CH * 22 * 32; e += 256) {
    int cc = e & 31;
    int rr = e >> 5;                             // 0..351
    int c  = rr / 22;
    int r  = rr - c * 22;
    int iy = py0 + r - 7, ix = px0 + cc - 7;
    float val = 0.f;
    if (iy >= 0 && iy < HW && ix >= 0 && ix < HW)
      val = xm[((size_t)img * CH + c) * HWHW + iy * HW + ix];
    patch[c][r][cc] = (f16)val;
  }
  __syncthreads();

  const v16h* Af = reinterpret_cast<const v16h*>(Asq1);
  v8f acc; for (int i = 0; i < 8; ++i) acc[i] = 0.f;
  for (int ks = 0; ks < KS2; ++ks) {
    v16h a = Af[ks * 32 + lane];                 // coalesced 32B/lane
    int chunk = ks * 2 + hi;                     // 0..239
    int c  = chunk / 15;
    int kh = chunk - c * 15;
    const f16* base = &patch[c][wid + kh][n];
    v16h bf;
#pragma unroll
    for (int u = 0; u < 15; ++u) bf[u] = base[u];
    bf[15] = (f16)0.f;
    acc = __builtin_amdgcn_wmma_f32_16x16x32_f16(
        false, a, false, bf, (short)0, acc, false, false);
  }

  const int py = py0 + wid;
  const int px = px0 + n;
  for (int r = 0; r < 8; ++r) {
    int ch = r + hi * 8;
    h1[((size_t)img * CH + ch) * HWHW + py * HW + px] = acc[r];
  }
}

// ---------------------------------------------------------------------------
// K3: sq2 conv (16->1) fused with 2x2 max-pool -> heatmaps (d_out[0:262144]).
// ---------------------------------------------------------------------------
__global__ void __launch_bounds__(256) k3_sq2_pool(
    const float* __restrict__ h1, const float* __restrict__ sq2_w,
    float* __restrict__ heat) {
  int g = blockIdx.x * 256 + threadIdx.x;
  if (g >= 64 * 64 * 64) return;
  int img = g >> 12;
  int rem = g & 4095;
  int pyp = rem >> 6, pxp = rem & 63;
  float acc[4] = {0.f, 0.f, 0.f, 0.f};
#pragma unroll 1
  for (int c = 0; c < CH; ++c) {
    const float* src = h1 + ((size_t)img * CH + c) * HWHW;
    const float* wp  = sq2_w + c * 225;
#pragma unroll 1
    for (int kh = 0; kh < 15; ++kh) {
      for (int kw = 0; kw < 15; ++kw) {
        float w = wp[kh * 15 + kw];
        for (int q = 0; q < 4; ++q) {
          int iy = 2 * pyp + (q >> 1) + kh - 7;
          int ix = 2 * pxp + (q & 1) + kw - 7;
          if (iy >= 0 && iy < HW && ix >= 0 && ix < HW)
            acc[q] = fmaf(w, src[iy * HW + ix], acc[q]);
        }
      }
    }
  }
  float mx = fmaxf(fmaxf(acc[0], acc[1]), fmaxf(acc[2], acc[3]));
  heat[(size_t)img * 4096 + pyp * 64 + pxp] = mx;
}

// ---------------------------------------------------------------------------
// K4: softmax over 64x64 + soft-argmax -> coords (d_out[262144:]).
// ---------------------------------------------------------------------------
__global__ void __launch_bounds__(256) k4_softargmax(
    const float* __restrict__ heat, float* __restrict__ coords) {
  __shared__ float red[256];
  const int img = blockIdx.x;
  const int tid = threadIdx.x;
  const float* h = heat + (size_t)img * 4096;

  float m = -3.402823466e38f;
  for (int i = tid; i < 4096; i += 256) m = fmaxf(m, h[i]);
  red[tid] = m; __syncthreads();
  for (int s = 128; s > 0; s >>= 1) {
    if (tid < s) red[tid] = fmaxf(red[tid], red[tid + s]);
    __syncthreads();
  }
  float gm = red[0]; __syncthreads();

  float se = 0.f, sx = 0.f, sy = 0.f;
  for (int i = tid; i < 4096; i += 256) {
    float e = __expf(h[i] - gm);
    se += e;
    sx += e * ((float)(i & 63) * (1.0f / 63.0f));
    sy += e * ((float)(i >> 6) * (1.0f / 63.0f));
  }
  red[tid] = se; __syncthreads();
  for (int s = 128; s > 0; s >>= 1) { if (tid < s) red[tid] += red[tid + s]; __syncthreads(); }
  float tse = red[0]; __syncthreads();
  red[tid] = sx; __syncthreads();
  for (int s = 128; s > 0; s >>= 1) { if (tid < s) red[tid] += red[tid + s]; __syncthreads(); }
  float tsx = red[0]; __syncthreads();
  red[tid] = sy; __syncthreads();
  for (int s = 128; s > 0; s >>= 1) { if (tid < s) red[tid] += red[tid + s]; __syncthreads(); }
  float tsy = red[0];

  if (tid == 0) {
    coords[img * 2 + 0] = tsx / tse;   // x = sum p * gx (col / 63)
    coords[img * 2 + 1] = tsy / tse;   // y = sum p * gy (row / 63)
  }
}

// ---------------------------------------------------------------------------
extern "C" void kernel_launch(void* const* d_in, const int* in_sizes, int n_in,
                              void* d_out, int out_size, void* d_ws, size_t ws_size,
                              hipStream_t stream) {
  (void)in_sizes; (void)n_in; (void)out_size; (void)ws_size;
  const float* x      = (const float*)d_in[0];   // (16,4,10,128,128)
  const float* head_w = (const float*)d_in[1];   // (16,2,15,15)
  const float* sq1_w  = (const float*)d_in[2];   // (16,16,15,15)
  const float* sq2_w  = (const float*)d_in[3];   // (1,16,15,15)
  const float* mtc    = (const float*)d_in[4];   // scalar

  float* out    = (float*)d_out;                 // heatmaps (262144) + coords (128)
  char*  ws     = (char*)d_ws;
  float* xm     = (float*)(ws);
  float* h1     = (float*)(ws + (size_t)67108864);
  f16*   Ahead  = (f16*)  (ws + (size_t)134217728);
  f16*   Asq1   = (f16*)  (ws + (size_t)134233088);
  float* coords = out + (size_t)64 * 4096;

  k0_pack      <<<dim3(17),      256, 0, stream>>>(head_w, sq1_w, Ahead, Asq1);
  k1_head_scan <<<dim3(128, 4),  256, 0, stream>>>(x, Ahead, mtc, xm);
  k2_sq1       <<<dim3(128, 64), 256, 0, stream>>>(xm, Asq1, h1);
  k3_sq2_pool  <<<dim3(1024),    256, 0, stream>>>(h1, sq2_w, out);
  k4_softargmax<<<dim3(64),      256, 0, stream>>>(out, coords);
}
